// CrossAttention_8375186227619
// MI455X (gfx1250) — matrix operand order repaired and checked by reading, not compiled, not run on hardware
//
#include <hip/hip_runtime.h>
#include <hip/hip_bf16.h>

typedef __bf16 bf16_t;
typedef __attribute__((ext_vector_type(16))) __bf16 v16bf;
typedef __attribute__((ext_vector_type(8)))  __bf16 v8bf;
typedef __attribute__((ext_vector_type(8)))  float  v8f;
typedef __attribute__((ext_vector_type(8)))  float  v8f32;
typedef __attribute__((ext_vector_type(4)))  int    v4i;

#define N_HEADS 8
#define D_HEAD  64
#define INNER   512
#define SQ      4096
#define SKV     4096
#define DM      512
#define DC      768
#define BATCH   2

union U16bf { v16bf v; v8bf h[2]; };

__device__ __forceinline__ v8bf cvt8(const float* p) {
  v8f32 t = *(const v8f32*)p;
  return __builtin_convertvector(t, v8bf);
}
__device__ __forceinline__ v8bf cvt8(const bf16_t* p) {
  return *(const v8bf*)p;
}
// A-fragment (16x32 bf16): lane row = lo, K chunks at [2kb..2kb+7], [2kb+8..2kb+15]
template <typename AT>
__device__ __forceinline__ v16bf loadAfrag(const AT* base, int kb) {
  U16bf u;
  u.h[0] = cvt8(base + 2 * kb);
  u.h[1] = cvt8(base + 2 * kb + 8);
  return u.v;
}

__device__ __forceinline__ v8f wmma_bf16(const v16bf& a, const v16bf& b, const v8f& c) {
  return __builtin_amdgcn_wmma_f32_16x16x32_bf16(false, a, false, b, (short)0, c, false, false);
}

// Row-wide (16-lane) max butterfly on VALU (v_permlane16_b32)
#if __has_builtin(__builtin_amdgcn_permlane16)
__device__ __forceinline__ float perm16(float v, unsigned s0, unsigned s1) {
  return __uint_as_float(__builtin_amdgcn_permlane16(
      __float_as_uint(v), __float_as_uint(v), s0, s1, false, false));
}
__device__ __forceinline__ float rowmax16(float v) {
  v = fmaxf(v, perm16(v, 0x67452301u, 0xefcdab89u));  // xor 1
  v = fmaxf(v, perm16(v, 0x54761032u, 0xdcfe98bau));  // xor 2
  v = fmaxf(v, perm16(v, 0x32107654u, 0xba98fedcu));  // xor 4
  v = fmaxf(v, perm16(v, 0xfedcba98u, 0x76543210u));  // xor 8
  return v;
}
#else
__device__ __forceinline__ float rowmax16(float v) {
  v = fmaxf(v, __shfl_xor(v, 1, 16));
  v = fmaxf(v, __shfl_xor(v, 2, 16));
  v = fmaxf(v, __shfl_xor(v, 4, 16));
  v = fmaxf(v, __shfl_xor(v, 8, 16));
  return v;
}
#endif

// ---- CDNA5 async global->LDS copy (ASYNCcnt-tracked), with sync fallback ----
#if __has_builtin(__builtin_amdgcn_global_load_async_to_lds_b128)
#define HAVE_ASYNC 1
#endif

__device__ __forceinline__ void copy16(const bf16_t* g, bf16_t* l) {
#ifdef HAVE_ASYNC
  __builtin_amdgcn_global_load_async_to_lds_b128(
      (__attribute__((address_space(1))) v4i*)(bf16_t*)g,
      (__attribute__((address_space(3))) v4i*)l, 0, 0);
#else
  *(v8bf*)l = *(const v8bf*)g;
#endif
}
template <int N>
__device__ __forceinline__ void wait_async() {
#ifdef HAVE_ASYNC
#if __has_builtin(__builtin_amdgcn_s_wait_asynccnt)
  __builtin_amdgcn_s_wait_asynccnt((unsigned short)N);
#else
  asm volatile("s_wait_asynccnt %0" :: "i"(N) : "memory");
#endif
#endif
}

// ---------------------------------------------------------------------------
// Weight transpose + convert: W[K][N] f32 -> Wt[N][K] bf16
// ---------------------------------------------------------------------------
__global__ __launch_bounds__(256)
void transpose_convert(const float* __restrict__ W, bf16_t* __restrict__ Wt,
                       int K, int N)
{
  int k = blockIdx.x * 32 + (threadIdx.x & 31);
  int n = blockIdx.y * 8 + (threadIdx.x >> 5);
  if (k < K && n < N)
    Wt[(long)n * K + k] = (bf16_t)W[(long)k * N + n];
}

// ---------------------------------------------------------------------------
// GEMM: C[M,N] = A[M,K] * Bt^T  (Bt is [N][K] bf16). One wave -> 16x64 tile.
// MODE 0: bf16 head-major [H][S_out][64], scaled by outScale  (Q,K proj)
// MODE 1: f32 (+bias) row-major                               (out proj)
// MODE 2: bf16 transposed head-major [H*64][S]                (V proj)
// ---------------------------------------------------------------------------
template <typename AT, int MODE>
__global__ __launch_bounds__(256)
void gemm_wmma(const AT* __restrict__ A, long strideA,
               const bf16_t* __restrict__ Bt,
               int M, int N, int K,
               bf16_t* __restrict__ outHM, long strideOutHM, int S_out,
               float outScale,
               float* __restrict__ outF, long strideOutF,
               const float* __restrict__ bias)
{
  const int lane = threadIdx.x;
  const int wid  = threadIdx.y;
  const int lo = lane & 15, hi = lane >> 4, kb = hi * 8;
  const int n0 = (blockIdx.x * 8 + wid) * 64;
  const int m0 = blockIdx.y * 16;
  const int bz = blockIdx.z;
  if (n0 >= N || m0 >= M) return;

  const AT*     Arow = A  + (long)bz * strideA + (long)(m0 + lo) * K;
  const bf16_t* Brow = Bt + (long)(n0 + lo) * K;

  v8f acc0 = {0.f,0.f,0.f,0.f,0.f,0.f,0.f,0.f};
  v8f acc1 = acc0, acc2 = acc0, acc3 = acc0;

  for (int k0 = 0; k0 < K; k0 += 32) {
    v16bf af = loadAfrag(Arow + k0, kb);
    const bf16_t* bp = Brow + k0 + 16 * hi;
    v16bf b0 = *(const v16bf*)(bp);
    v16bf b1 = *(const v16bf*)(bp + 16 * (long)K);
    v16bf b2 = *(const v16bf*)(bp + 32 * (long)K);
    v16bf b3 = *(const v16bf*)(bp + 48 * (long)K);
    acc0 = wmma_bf16(af, b0, acc0);
    acc1 = wmma_bf16(af, b1, acc1);
    acc2 = wmma_bf16(af, b2, acc2);
    acc3 = wmma_bf16(af, b3, acc3);
  }

  v8f accs[4] = {acc0, acc1, acc2, acc3};
#pragma unroll
  for (int t = 0; t < 4; ++t) {
#pragma unroll
    for (int r = 0; r < 8; ++r) {
      int m = m0 + r + 8 * hi;
      int n = n0 + t * 16 + lo;
      if (MODE == 0) {
        int h = n >> 6, d = n & 63;
        outHM[(long)bz * strideOutHM + ((long)h * S_out + m) * D_HEAD + d] =
            (bf16_t)(accs[t][r] * outScale);
      } else if (MODE == 2) {
        outHM[(long)bz * strideOutHM + (long)n * S_out + m] = (bf16_t)accs[t][r];
      } else {
        outF[(long)bz * strideOutF + (long)m * N + n] = accs[t][r] + bias[n];
      }
    }
  }
}

// ---------------------------------------------------------------------------
// Flash attention. Block = 8 waves sharing one (b,h); K/V tiles (32 kv rows)
// are staged into LDS once per block via async global->LDS copies, double-
// buffered. Each wave owns 32 q rows (2 sub-tiles). Q pre-scaled by
// 0.125*log2(e): softmax in exp2 domain. Row-sums via P*ones WMMA; row-max
// via permlane16 butterfly. 9 WMMAs per sub-tile per 32 kv rows.
// ---------------------------------------------------------------------------
#define QSUB 2
#define KVIT (SKV / 32)

__global__ __launch_bounds__(256)
void flash_attn_wmma(const bf16_t* __restrict__ Q,
                     const bf16_t* __restrict__ Kk,
                     const bf16_t* __restrict__ Vt,
                     bf16_t* __restrict__ attn)
{
  __shared__ bf16_t Klds[2][32][72];  // 144B rows: aligned, conflict-free
  __shared__ bf16_t Vlds[2][64][40];  // 80B rows
  __shared__ bf16_t Plds[8][16][40];  // per-wave P relayout tile

  const int lane = threadIdx.x, wid = threadIdx.y;
  const int tid  = wid * 32 + lane;
  const int lo = lane & 15, hi = lane >> 4, kb = hi * 8;
  const int q0 = (blockIdx.x * 8 + wid) * (16 * QSUB);
  const int h  = blockIdx.y, b = blockIdx.z;

  const long bh = (long)b * N_HEADS + h;
  const bf16_t* Qb  = Q  + bh * (long)SQ  * D_HEAD;
  const bf16_t* Kb  = Kk + bh * (long)SKV * D_HEAD;
  const bf16_t* Vtb = Vt + bh * (long)D_HEAD * SKV;

  // per-thread copy slots: K tile 32x(8x16B), V tile 64x(4x16B)
  const int krow = tid >> 3, kc8 = (tid & 7) * 8;
  const int vrow = tid >> 2, vc8 = (tid & 3) * 8;

  // Q A-fragments (pre-scaled in projection)
  v16bf qa[QSUB][2];
#pragma unroll
  for (int s = 0; s < QSUB; ++s) {
    const bf16_t* Qrow = Qb + (long)(q0 + s * 16 + lo) * D_HEAD;
    qa[s][0] = loadAfrag(Qrow, kb);
    qa[s][1] = loadAfrag(Qrow + 32, kb);
  }

  v16bf ones;
#pragma unroll
  for (int j = 0; j < 16; ++j) ones[j] = (bf16_t)1.0f;

  float mst[QSUB][8];
  v8f o[QSUB][4], osum[QSUB];
  const v8f vz = {0.f,0.f,0.f,0.f,0.f,0.f,0.f,0.f};
#pragma unroll
  for (int s = 0; s < QSUB; ++s) {
    osum[s] = vz;
#pragma unroll
    for (int t = 0; t < 4; ++t) o[s][t] = vz;
#pragma unroll
    for (int r = 0; r < 8; ++r) mst[s][r] = -1e30f;
  }

  // prologue: stage tile 0
  copy16(Kb + (long)krow * D_HEAD + kc8, &Klds[0][krow][kc8]);
  copy16(Vtb + (long)vrow * SKV + vc8,   &Vlds[0][vrow][vc8]);

  for (int it = 0; it < KVIT; ++it) {
    const int buf = it & 1;
    __syncthreads();                       // prev readers of buf^1 done
    if (it + 1 < KVIT) {                   // stage next tile into buf^1
      const long kvn = (long)(it + 1) * 32;
      copy16(Kb + (kvn + krow) * D_HEAD + kc8, &Klds[buf ^ 1][krow][kc8]);
      copy16(Vtb + (long)vrow * SKV + kvn + vc8, &Vlds[buf ^ 1][vrow][vc8]);
      wait_async<2>();                     // our copies for THIS tile done
    } else {
      wait_async<0>();
    }
    __syncthreads();                       // tile visible to all waves

    // K^T B-fragments from LDS (shared across waves & sub-tiles)
    v16bf kA0 = *(const v16bf*)(&Klds[buf][lo][16 * hi]);
    v16bf kA1 = *(const v16bf*)(&Klds[buf][lo][32 + 16 * hi]);
    v16bf kB0 = *(const v16bf*)(&Klds[buf][16 + lo][16 * hi]);
    v16bf kB1 = *(const v16bf*)(&Klds[buf][16 + lo][32 + 16 * hi]);

    // V B-fragments from LDS
    v16bf vb0 = *(const v16bf*)(&Vlds[buf][lo][16 * hi]);
    v16bf vb1 = *(const v16bf*)(&Vlds[buf][16 + lo][16 * hi]);
    v16bf vb2 = *(const v16bf*)(&Vlds[buf][32 + lo][16 * hi]);
    v16bf vb3 = *(const v16bf*)(&Vlds[buf][48 + lo][16 * hi]);

#pragma unroll
    for (int s = 0; s < QSUB; ++s) {
      v8f s0 = vz, s1 = vz;
      s0 = wmma_bf16(qa[s][0], kA0, s0);
      s0 = wmma_bf16(qa[s][1], kA1, s0);
      s1 = wmma_bf16(qa[s][0], kB0, s1);
      s1 = wmma_bf16(qa[s][1], kB1, s1);

      float alpha[8];
#pragma unroll
      for (int r = 0; r < 8; ++r) {
        float v0 = s0[r], v1 = s1[r];
        float mx = rowmax16(fmaxf(v0, v1));
        float mnew = fmaxf(mst[s][r], mx);
        float al = __builtin_amdgcn_exp2f(mst[s][r] - mnew);
        float p0 = __builtin_amdgcn_exp2f(v0 - mnew);
        float p1 = __builtin_amdgcn_exp2f(v1 - mnew);
        mst[s][r] = mnew;
        alpha[r] = al;
        int m = r + 8 * hi;
        Plds[wid][m][lo]      = (bf16_t)p0;
        Plds[wid][m][16 + lo] = (bf16_t)p1;
      }

#pragma unroll
      for (int r = 0; r < 8; ++r) {
        o[s][0][r] *= alpha[r]; o[s][1][r] *= alpha[r];
        o[s][2][r] *= alpha[r]; o[s][3][r] *= alpha[r];
        osum[s][r] *= alpha[r];
      }

      v16bf pa = loadAfrag(&Plds[wid][lo][0], kb);

      o[s][0] = wmma_bf16(pa, vb0, o[s][0]);
      o[s][1] = wmma_bf16(pa, vb1, o[s][1]);
      o[s][2] = wmma_bf16(pa, vb2, o[s][2]);
      o[s][3] = wmma_bf16(pa, vb3, o[s][3]);
      osum[s] = wmma_bf16(pa, ones, osum[s]);
    }
  }

  // normalize + store to [B][Sq][H*64] bf16
#pragma unroll
  for (int s = 0; s < QSUB; ++s) {
#pragma unroll
    for (int r = 0; r < 8; ++r) {
      float inv = __builtin_amdgcn_rcpf(osum[s][r]);
      int m = q0 + s * 16 + r + 8 * hi;
      long base = ((long)b * SQ + m) * INNER + (long)h * D_HEAD + lo;
      attn[base]      = (bf16_t)(o[s][0][r] * inv);
      attn[base + 16] = (bf16_t)(o[s][1][r] * inv);
      attn[base + 32] = (bf16_t)(o[s][2][r] * inv);
      attn[base + 48] = (bf16_t)(o[s][3][r] * inv);
    }
  }
}

// ---------------------------------------------------------------------------
extern "C" void kernel_launch(void* const* d_in, const int* in_sizes, int n_in,
                              void* d_out, int out_size, void* d_ws, size_t ws_size,
                              hipStream_t stream) {
  const float* x   = (const float*)d_in[0];
  const float* ctx = (const float*)d_in[1];
  const float* Wq  = (const float*)d_in[2];
  const float* Wk  = (const float*)d_in[3];
  const float* Wv  = (const float*)d_in[4];
  const float* Wo  = (const float*)d_in[5];
  const float* bo  = (const float*)d_in[6];
  float* out = (float*)d_out;

  const size_t perT = (size_t)BATCH * N_HEADS * SQ * D_HEAD;
  bf16_t* Qws  = (bf16_t*)d_ws;
  bf16_t* Kws  = Qws + perT;
  bf16_t* Vtws = Kws + perT;
  bf16_t* Attn = Vtws + perT;
  bf16_t* WqT  = Attn + perT;
  bf16_t* WkT  = WqT + (size_t)INNER * DM;
  bf16_t* WvT  = WkT + (size_t)INNER * DC;
  bf16_t* WoT  = WvT + (size_t)INNER * DC;

  dim3 blkT(256);
  transpose_convert<<<dim3(DM / 32, INNER / 8), blkT, 0, stream>>>(Wq, WqT, DM, INNER);
  transpose_convert<<<dim3(DC / 32, INNER / 8), blkT, 0, stream>>>(Wk, WkT, DC, INNER);
  transpose_convert<<<dim3(DC / 32, INNER / 8), blkT, 0, stream>>>(Wv, WvT, DC, INNER);
  transpose_convert<<<dim3(INNER / 32, INNER / 8), blkT, 0, stream>>>(Wo, WoT, INNER, INNER);

  dim3 blk(32, 8);
  dim3 gP(1, SQ / 16, BATCH);
  const long hmStride = (long)N_HEADS * SQ * D_HEAD;
  const float qScale = 0.125f * 1.44269504088896340736f;  // 1/sqrt(64)*log2(e)

  gemm_wmma<float, 0><<<gP, blk, 0, stream>>>(
      x, (long)SQ * DM, WqT, SQ, INNER, DM,
      Qws, hmStride, SQ, qScale, nullptr, 0, nullptr);
  gemm_wmma<float, 0><<<gP, blk, 0, stream>>>(
      ctx, (long)SKV * DC, WkT, SKV, INNER, DC,
      Kws, hmStride, SKV, 1.0f, nullptr, 0, nullptr);
  gemm_wmma<float, 2><<<gP, blk, 0, stream>>>(
      ctx, (long)SKV * DC, WvT, SKV, INNER, DC,
      Vtws, hmStride, SKV, 1.0f, nullptr, 0, nullptr);

  dim3 gF(SQ / 256, N_HEADS, BATCH);
  flash_attn_wmma<<<gF, blk, 0, stream>>>(Qws, Kws, Vtws, Attn);

  gemm_wmma<bf16_t, 1><<<gP, blk, 0, stream>>>(
      Attn, (long)SQ * INNER, WoT, SQ, INNER, INNER,
      nullptr, 0, 0, 1.0f, out, (long)SQ * INNER, bo);
}
